// FusedMistralLayerBaseline_80908593922052
// MI455X (gfx1250) — matrix operand order repaired and checked by reading, not compiled, not run on hardware
//
#include <hip/hip_runtime.h>
#include <hip/hip_bf16.h>

// ---------------------------------------------------------------------------
// Fused Mistral layer for gfx1250 (MI455X), wave32 + WMMA bf16.
// All GEMMs use v_wmma_f32_16x16x32_bf16 (fp32 accumulate).
// GEMM is M-register-blocked: one weight (B) fragment feeds 8 WMMAs.
// ---------------------------------------------------------------------------

typedef __attribute__((ext_vector_type(16))) __bf16 bf16x16;
typedef __attribute__((ext_vector_type(8)))  __bf16 bf16x8;
typedef __attribute__((ext_vector_type(8)))  float  f32x8;
typedef __attribute__((ext_vector_type(4)))  float  f32x4;

#define CAT16(lo, hi) __builtin_shufflevector((lo), (hi), 0,1,2,3,4,5,6,7,8,9,10,11,12,13,14,15)

__device__ __forceinline__ f32x8 wmma_bf16(bf16x16 a, bf16x16 b, f32x8 c) {
  // 8 args: (neg_a, A, neg_b, B, c_mod, C, reuse_a, reuse_b)
  return __builtin_amdgcn_wmma_f32_16x16x32_bf16(false, a, false, b, (short)0, c,
                                                 false, false);
}

// Layer constants (match reference)
#define S_  1024
#define D_  4096
#define H_  32
#define HD_ 128
#define KVH_ 8
#define KVD_ 1024      // KVH*HD
#define FF_ 14336
#define R_  16

// ---------------------------------------------------------------------------
// RMSNorm: one block per row; writes fp32 and bf16 copies.
// ---------------------------------------------------------------------------
__global__ __launch_bounds__(256)
void rmsnorm_kernel(const float* __restrict__ X, const float* __restrict__ w,
                    float* __restrict__ Yf, __bf16* __restrict__ Yb, int D) {
  __shared__ float red[8];
  const int s = blockIdx.x;
  const float* xrow = X + (size_t)s * D;
  float ss = 0.f;
  for (int d = threadIdx.x; d < D; d += 256) { float v = xrow[d]; ss += v * v; }
  for (int m = 16; m >= 1; m >>= 1) ss += __shfl_xor(ss, m, 32);
  if ((threadIdx.x & 31) == 0) red[threadIdx.x >> 5] = ss;
  __syncthreads();
  float tot = 0.f;
#pragma unroll
  for (int i = 0; i < 8; ++i) tot += red[i];
  const float scale = rsqrtf(tot / (float)D + 1e-5f);
  for (int d = threadIdx.x; d < D; d += 256) {
    float v = xrow[d] * scale * w[d];
    Yf[(size_t)s * D + d] = v;
    Yb[(size_t)s * D + d] = (__bf16)v;
  }
}

// ---------------------------------------------------------------------------
// LoRA first stage: tmp[s, r] = sum_d X[s,d] * la[d,r]   (r = 0..15)
// ---------------------------------------------------------------------------
__global__ __launch_bounds__(256)
void lora_xa_kernel(const float* __restrict__ X, const float* __restrict__ la,
                    float* __restrict__ tmp, int Din) {
  __shared__ float red[256];
  const int s = blockIdx.x, t = threadIdx.x;
  const int r = t & 15, g = t >> 4;                 // 16 groups of 16
  const float* xrow = X + (size_t)s * Din;
  float acc = 0.f;
  for (int d = g; d < Din; d += 16) acc += xrow[d] * la[(size_t)d * R_ + r];
  red[t] = acc;
  __syncthreads();
  if (g == 0) {
    float v = 0.f;
#pragma unroll
    for (int gg = 0; gg < 16; ++gg) v += red[gg * 16 + r];
    tmp[(size_t)s * R_ + r] = v;
  }
}

// ---------------------------------------------------------------------------
// WMMA GEMM: C[M,N] = A_bf16[M,K] @ W_f32[N,K]^T  (+ tmp@lb LoRA) (+ resid)
// Block = 8 waves, each wave owns one 16-col N strip and MT=8 stacked M tiles:
// one converted weight fragment is reused by 8 WMMAs (weight stream is the
// bottleneck on MI455X: ~873 MB fp32 @ 23.3 TB/s; re-read factor drops to
// M/128 = 8, and wq/wk/wv/wo strips then live in the 192 MB L2).
// Block tile = 128(M) x 128(N). A and B fragments both use contiguous k order (k = 16*h + s).
// ---------------------------------------------------------------------------
#define MT_ 8
__global__ __launch_bounds__(256)
void gemm_wmma_kernel(const __bf16* __restrict__ A, const float* __restrict__ W,
                      const float* __restrict__ lora_tmp, const float* __restrict__ lb,
                      const float* __restrict__ resid, float* __restrict__ C,
                      int M, int N, int K) {
  const int lane = threadIdx.x & 31;
  const int wave = threadIdx.x >> 5;
  const int m0 = blockIdx.y * (MT_ * 16);
  const int n0 = blockIdx.x * 128 + wave * 16;
  const int hl = lane >> 4, l15 = lane & 15;

  const __bf16* arow = A + (size_t)(m0 + l15) * K;   // tile t adds t*16*K
  const float*  wrow = W + (size_t)(n0 + l15) * K;

  f32x8 acc[MT_];
#pragma unroll
  for (int t = 0; t < MT_; ++t) acc[t] = {};

  for (int k = 0; k < K; k += 32) {
    __builtin_prefetch(wrow + k + 256, 0, 1);   // global_prefetch on weight stream
    // B fragment 32x16: B[k][n] = W[n][k]; 16 contiguous fp32 -> bf16 (once)
    const f32x4* wp = (const f32x4*)(wrow + k + hl * 16);
    bf16x16 bfrag;
#pragma unroll
    for (int q4 = 0; q4 < 4; ++q4) {
      f32x4 wv = wp[q4];
      bfrag[q4 * 4 + 0] = (__bf16)wv[0];
      bfrag[q4 * 4 + 1] = (__bf16)wv[1];
      bfrag[q4 * 4 + 2] = (__bf16)wv[2];
      bfrag[q4 * 4 + 3] = (__bf16)wv[3];
    }
    // 8 stacked A fragments (16x32 bf16 each), two contiguous 16B loads per lane
#pragma unroll
    for (int t = 0; t < MT_; ++t) {
      const __bf16* at = arow + (size_t)t * 16 * K;
      bf16x8 a0 = *(const bf16x8*)(at + k + hl * 16);
      bf16x8 a1 = *(const bf16x8*)(at + k + hl * 16 + 8);
      acc[t] = wmma_bf16(CAT16(a0, a1), bfrag, acc[t]);
    }
  }

  const int nn = n0 + l15;
  // LoRA second stage: lb column for this lane is loop-invariant -> hoist
  float lbv[R_];
  if (lora_tmp) {
#pragma unroll
    for (int j = 0; j < R_; ++j) lbv[j] = lb[(size_t)j * N + nn];
  }
#pragma unroll
  for (int t = 0; t < MT_; ++t) {
#pragma unroll
    for (int r = 0; r < 8; ++r) {
      const int m = m0 + t * 16 + hl * 8 + r;
      float v = acc[t][r];
      if (lora_tmp) {
        const float* tp = lora_tmp + (size_t)m * R_;
#pragma unroll
        for (int j = 0; j < R_; ++j) v += tp[j] * lbv[j];
      }
      if (resid) v += resid[(size_t)m * N + nn];
      C[(size_t)m * N + nn] = v;
    }
  }
}

// ---------------------------------------------------------------------------
// RoPE (rotate_half) + convert to bf16. grid (S, nHeads), 64 threads.
// ---------------------------------------------------------------------------
__global__ __launch_bounds__(64)
void rope_kernel(const float* __restrict__ Q, const float* __restrict__ cosT,
                 const float* __restrict__ sinT, __bf16* __restrict__ Qb, int nHeads) {
  const int s = blockIdx.x, h = blockIdx.y, i = threadIdx.x;   // i in [0,64)
  const float* base = Q + ((size_t)s * nHeads + h) * HD_;
  __bf16* ob = Qb + ((size_t)s * nHeads + h) * HD_;
  float x1 = base[i], x2 = base[i + 64];
  float c1 = cosT[(size_t)s * HD_ + i],      s1 = sinT[(size_t)s * HD_ + i];
  float c2 = cosT[(size_t)s * HD_ + i + 64], s2 = sinT[(size_t)s * HD_ + i + 64];
  ob[i]      = (__bf16)(x1 * c1 - x2 * s1);
  ob[i + 64] = (__bf16)(x2 * c2 + x1 * s2);
}

__global__ __launch_bounds__(256)
void f32_to_bf16_kernel(const float* __restrict__ X, __bf16* __restrict__ Y, size_t n) {
  size_t i = (size_t)blockIdx.x * 256 + threadIdx.x;
  if (i < n) Y[i] = (__bf16)X[i];
}

// ---------------------------------------------------------------------------
// Flash attention (causal, GQA 4:1). One wave per (head, 16-row q tile).
// QK^T and P*V both through bf16 WMMA; online softmax with half-wave shuffles.
// ---------------------------------------------------------------------------
__global__ __launch_bounds__(256)
void attn_kernel(const __bf16* __restrict__ Qb, const __bf16* __restrict__ Kb,
                 const __bf16* __restrict__ Vb, float* __restrict__ Of,
                 __bf16* __restrict__ Ob) {
  __shared__ __bf16 pbuf[8][16 * 32];       // per-wave P tile (C-layout -> A-layout)
  const int lane = threadIdx.x & 31;
  const int wave = threadIdx.x >> 5;
  const int hl = lane >> 4, l15 = lane & 15;
  const int qt = blockIdx.x * 8 + wave;     // q tile 0..63
  const int h  = blockIdx.y;                // head 0..31
  const int kvh = h >> 2;                   // GQA: H/KVH = 4
  const int q0 = qt * 16;
  const float scale = 0.08838834764831845f; // 1/sqrt(128)

  // Q A-fragments resident: HD=128 -> 4 chunks of K=32
  bf16x16 qfrag[4];
  const __bf16* qrow = Qb + (size_t)(q0 + l15) * D_ + (size_t)h * HD_;
#pragma unroll
  for (int ks = 0; ks < 4; ++ks) {
    bf16x8 a0 = *(const bf16x8*)(qrow + ks * 32 + hl * 16);
    bf16x8 a1 = *(const bf16x8*)(qrow + ks * 32 + hl * 16 + 8);
    qfrag[ks] = CAT16(a0, a1);
  }

  f32x8 acc[8];
  float mrun[8], lrun[8];
#pragma unroll
  for (int c = 0; c < 8; ++c) acc[c] = {};
#pragma unroll
  for (int r = 0; r < 8; ++r) { mrun[r] = -3.0e38f; lrun[r] = 0.f; }

  __bf16* pb = &pbuf[wave][0];
  const int kv_hi = q0 + 16;
  for (int kv = 0; kv < kv_hi; kv += 32) {
    // ---- scores: two 16x16 tiles (keys kv..kv+15, kv+16..kv+31)
    f32x8 sc[2];
#pragma unroll
    for (int t = 0; t < 2; ++t) {
      f32x8 c = {};
      const int kt = kv + t * 16;
      const __bf16* krow = Kb + (size_t)(kt + l15) * KVD_ + (size_t)kvh * HD_;
#pragma unroll
      for (int ks = 0; ks < 4; ++ks) {
        bf16x8 b0 = *(const bf16x8*)(krow + ks * 32 + hl * 16);
        bf16x8 b1 = *(const bf16x8*)(krow + ks * 32 + hl * 16 + 8);
        c = wmma_bf16(qfrag[ks], CAT16(b0, b1), c);
      }
      sc[t] = c;
    }
    // ---- online softmax per row (rows hl*8+r live in this half's lanes)
#pragma unroll
    for (int r = 0; r < 8; ++r) {
      const int mloc = hl * 8 + r;
      const int qidx = q0 + mloc;
      float s0 = (kv + l15      <= qidx) ? sc[0][r] * scale : -1e30f;
      float s1 = (kv + 16 + l15 <= qidx) ? sc[1][r] * scale : -1e30f;
      float mx = fmaxf(s0, s1);
#pragma unroll
      for (int m = 8; m >= 1; m >>= 1) mx = fmaxf(mx, __shfl_xor(mx, m, 32));
      const float mnew = fmaxf(mrun[r], mx);
      const float p0 = __expf(s0 - mnew), p1 = __expf(s1 - mnew);
      pb[mloc * 32 + l15]      = (__bf16)p0;
      pb[mloc * 32 + 16 + l15] = (__bf16)p1;
      float rs = p0 + p1;
#pragma unroll
      for (int m = 8; m >= 1; m >>= 1) rs += __shfl_xor(rs, m, 32);
      const float corr = __expf(mrun[r] - mnew);
      lrun[r] = lrun[r] * corr + rs;
      mrun[r] = mnew;
#pragma unroll
      for (int c = 0; c < 8; ++c) acc[c][r] *= corr;
    }
    asm volatile("s_wait_dscnt 0" ::: "memory");
    // ---- reload P in A-layout (16x32) from LDS
    const __bf16* pl = pb + l15 * 32;
    bf16x8 p0v = *(const bf16x8*)(pl + hl * 16);
    bf16x8 p1v = *(const bf16x8*)(pl + hl * 16 + 8);
    bf16x16 pa = CAT16(p0v, p1v);
    asm volatile("" ::: "memory");
    // ---- O += P @ V  (8 column groups of 16 over HD=128)
#pragma unroll
    for (int c = 0; c < 8; ++c) {
      bf16x16 vf;
      const __bf16* vcol = Vb + (size_t)(kv + hl * 16) * KVD_ + (size_t)kvh * HD_ + c * 16 + l15;
#pragma unroll
      for (int j = 0; j < 16; ++j) vf[j] = vcol[(size_t)j * KVD_];
      acc[c] = wmma_bf16(pa, vf, acc[c]);
    }
  }
  // ---- epilogue: normalize by row sum, write fp32 + bf16
#pragma unroll
  for (int r = 0; r < 8; ++r) {
    const int m = q0 + hl * 8 + r;
    const float inv = 1.f / lrun[r];
#pragma unroll
    for (int c = 0; c < 8; ++c) {
      const float v = acc[c][r] * inv;
      const size_t idx = (size_t)m * D_ + (size_t)h * HD_ + c * 16 + l15;
      Of[idx] = v;
      Ob[idx] = (__bf16)v;
    }
  }
}

// ---------------------------------------------------------------------------
// h = silu(gate) * up ; writes fp32 (in place over gate) + bf16
// ---------------------------------------------------------------------------
__global__ __launch_bounds__(256)
void silu_mul_kernel(float* __restrict__ gate, const float* __restrict__ up,
                     __bf16* __restrict__ hb, size_t n) {
  size_t i = (size_t)blockIdx.x * 256 + threadIdx.x;
  if (i < n) {
    float g = gate[i];
    float v = (g / (1.f + __expf(-g))) * up[i];
    gate[i] = v;
    hb[i] = (__bf16)v;
  }
}

// ---------------------------------------------------------------------------
// Host launch
// ---------------------------------------------------------------------------
extern "C" void kernel_launch(void* const* d_in, const int* in_sizes, int n_in,
                              void* d_out, int out_size, void* d_ws, size_t ws_size,
                              hipStream_t stream) {
  const float* x    = (const float*)d_in[0];
  const float* nw1  = (const float*)d_in[1];
  const float* nw2  = (const float*)d_in[2];
  const float* cosT = (const float*)d_in[3];
  const float* sinT = (const float*)d_in[4];
  // d_in[5] = mask (causal handled analytically)
  const float* wq = (const float*)d_in[6],  *la_q = (const float*)d_in[7],  *lb_q = (const float*)d_in[8];
  const float* wk = (const float*)d_in[9],  *la_k = (const float*)d_in[10], *lb_k = (const float*)d_in[11];
  const float* wv = (const float*)d_in[12], *la_v = (const float*)d_in[13], *lb_v = (const float*)d_in[14];
  const float* wo = (const float*)d_in[15], *la_o = (const float*)d_in[16], *lb_o = (const float*)d_in[17];
  const float* wg = (const float*)d_in[18], *la_g = (const float*)d_in[19], *lb_g = (const float*)d_in[20];
  const float* wu = (const float*)d_in[21], *la_u = (const float*)d_in[22], *lb_u = (const float*)d_in[23];
  const float* wd = (const float*)d_in[24], *la_d = (const float*)d_in[25], *lb_d = (const float*)d_in[26];
  float* out = (float*)d_out;

  // workspace carve-out (256B aligned blocks)
  char* base = (char*)d_ws;
  size_t off = 0;
  auto carve = [&](size_t bytes) -> char* {
    char* p = base + off;
    off = (off + bytes + 255) & ~(size_t)255;
    return p;
  };
  float*  xn1_f = (float*)carve((size_t)S_ * D_ * 4);
  __bf16* xn1_b = (__bf16*)carve((size_t)S_ * D_ * 2);
  float*  tmp   = (float*)carve((size_t)S_ * R_ * 4);
  float*  q_f   = (float*)carve((size_t)S_ * D_ * 4);
  float*  k_f   = (float*)carve((size_t)S_ * KVD_ * 4);
  float*  v_f   = (float*)carve((size_t)S_ * KVD_ * 4);
  __bf16* q_b   = (__bf16*)carve((size_t)S_ * D_ * 2);
  __bf16* k_b   = (__bf16*)carve((size_t)S_ * KVD_ * 2);
  __bf16* v_b   = (__bf16*)carve((size_t)S_ * KVD_ * 2);
  float*  at_f  = (float*)carve((size_t)S_ * D_ * 4);
  __bf16* at_b  = (__bf16*)carve((size_t)S_ * D_ * 2);
  float*  xmed  = (float*)carve((size_t)S_ * D_ * 4);
  float*  xn2_f = (float*)carve((size_t)S_ * D_ * 4);
  __bf16* xn2_b = (__bf16*)carve((size_t)S_ * D_ * 2);
  float*  gate  = (float*)carve((size_t)S_ * FF_ * 4);   // reused as h_f32
  float*  up    = (float*)carve((size_t)S_ * FF_ * 4);
  __bf16* h_b   = (__bf16*)carve((size_t)S_ * FF_ * 2);
  (void)ws_size; (void)in_sizes; (void)n_in; (void)out_size;

  dim3 blk256(256);
  auto gemm_grid = [](int M, int N) { return dim3(N / 128, M / (MT_ * 16)); };

  // --- attention half ---
  rmsnorm_kernel<<<S_, blk256, 0, stream>>>(x, nw1, xn1_f, xn1_b, D_);

  lora_xa_kernel<<<S_, blk256, 0, stream>>>(xn1_f, la_q, tmp, D_);
  gemm_wmma_kernel<<<gemm_grid(S_, D_), blk256, 0, stream>>>(
      xn1_b, wq, tmp, lb_q, nullptr, q_f, S_, D_, D_);
  lora_xa_kernel<<<S_, blk256, 0, stream>>>(xn1_f, la_k, tmp, D_);
  gemm_wmma_kernel<<<gemm_grid(S_, KVD_), blk256, 0, stream>>>(
      xn1_b, wk, tmp, lb_k, nullptr, k_f, S_, KVD_, D_);
  lora_xa_kernel<<<S_, blk256, 0, stream>>>(xn1_f, la_v, tmp, D_);
  gemm_wmma_kernel<<<gemm_grid(S_, KVD_), blk256, 0, stream>>>(
      xn1_b, wv, tmp, lb_v, nullptr, v_f, S_, KVD_, D_);

  rope_kernel<<<dim3(S_, H_),  dim3(64), 0, stream>>>(q_f, cosT, sinT, q_b, H_);
  rope_kernel<<<dim3(S_, KVH_), dim3(64), 0, stream>>>(k_f, cosT, sinT, k_b, KVH_);
  {
    size_t n = (size_t)S_ * KVD_;
    f32_to_bf16_kernel<<<(unsigned)((n + 255) / 256), blk256, 0, stream>>>(v_f, v_b, n);
  }

  attn_kernel<<<dim3(S_ / 16 / 8, H_), blk256, 0, stream>>>(q_b, k_b, v_b, at_f, at_b);

  lora_xa_kernel<<<S_, blk256, 0, stream>>>(at_f, la_o, tmp, D_);
  gemm_wmma_kernel<<<gemm_grid(S_, D_), blk256, 0, stream>>>(
      at_b, wo, tmp, lb_o, /*resid=*/x, xmed, S_, D_, D_);

  // --- MLP half ---
  rmsnorm_kernel<<<S_, blk256, 0, stream>>>(xmed, nw2, xn2_f, xn2_b, D_);

  lora_xa_kernel<<<S_, blk256, 0, stream>>>(xn2_f, la_g, tmp, D_);
  gemm_wmma_kernel<<<gemm_grid(S_, FF_), blk256, 0, stream>>>(
      xn2_b, wg, tmp, lb_g, nullptr, gate, S_, FF_, D_);
  lora_xa_kernel<<<S_, blk256, 0, stream>>>(xn2_f, la_u, tmp, D_);
  gemm_wmma_kernel<<<gemm_grid(S_, FF_), blk256, 0, stream>>>(
      xn2_b, wu, tmp, lb_u, nullptr, up, S_, FF_, D_);

  {
    size_t n = (size_t)S_ * FF_;
    silu_mul_kernel<<<(unsigned)((n + 255) / 256), blk256, 0, stream>>>(gate, up, h_b, n);
  }

  lora_xa_kernel<<<S_, blk256, 0, stream>>>(gate /*h_f32*/, la_d, tmp, FF_);
  gemm_wmma_kernel<<<gemm_grid(S_, D_), blk256, 0, stream>>>(
      h_b, wd, tmp, lb_d, /*resid=*/xmed, out, S_, D_, FF_);
}
